// IBR_30142080483483
// MI455X (gfx1250) — compile-verified, matrix operand order checked by reading, not run on hardware
//
#include <hip/hip_runtime.h>
#include <hip/hip_bf16.h>

// ---------------- problem constants (from reference) ----------------
#define NV    8       // views
#define IH    256     // image H
#define IW    256     // image W
#define LF    32      // feature half-channels
#define HID   64      // hidden dim
#define NC    64      // coarse samples
#define NSF   128     // fine samples (NC+NF)
#define NR    1024    // rays
#define KPAD  64      // padded input dim (41 -> 64)

typedef __attribute__((ext_vector_type(16))) _Float16 v16h;
typedef __attribute__((ext_vector_type(8)))  _Float16 v8h;
typedef __attribute__((ext_vector_type(8)))  float    v8f;

// ------------------------------------------------------------------
// Prep: P[v] = K[v] @ E[v]  (3x3 @ 3x4)
// ------------------------------------------------------------------
__global__ void ibr_prep_pmat(const float* __restrict__ Kin,
                              const float* __restrict__ Ein,
                              float* __restrict__ Pm) {
  int t = threadIdx.x;
  if (t >= NV * 12) return;
  int v = t / 12, rem = t % 12, i = rem / 4, j = rem % 4;
  float acc = 0.f;
  for (int k = 0; k < 3; ++k)
    acc += Kin[v * 9 + i * 3 + k] * Ein[v * 12 + k * 4 + j];
  Pm[v * 12 + i * 4 + j] = acc;
}

// Prep: transpose + zero-pad weights to f16 [n][KPAD], Wt[n][k] = W[k][n]
__global__ void ibr_prep_w(const float* __restrict__ src,
                           _Float16* __restrict__ dst, int Kdim) {
  int idx = blockIdx.x * blockDim.x + threadIdx.x;
  if (idx >= HID * KPAD) return;
  int n = idx >> 6, k = idx & 63;
  dst[n * KPAD + k] = (k < Kdim) ? (_Float16)src[k * HID + n] : (_Float16)0.0f;
}

// ------------------------------------------------------------------
// Sampling: one thread per (point, view) row.
// Builds the padded 41->64 f16 input row, raw rgb (f32), mask coverage.
// ------------------------------------------------------------------
__global__ void ibr_sample(const float* __restrict__ ray,
                           const float* __restrict__ Pm,
                           const float* __restrict__ shape,
                           const float* __restrict__ campos,
                           const float* __restrict__ maskT,
                           const float* __restrict__ imgT,
                           const float* __restrict__ featT,
                           const float* __restrict__ fulld, // depths (fine) or null (coarse)
                           int NS, int fs,
                           _Float16* __restrict__ X,
                           float* __restrict__ rawrgb,
                           float* __restrict__ maskv) {
  int row = blockIdx.x * blockDim.x + threadIdx.x;
  int total = NR * NS * NV;
  if (row >= total) return;
  int v = row & 7;
  int p = row >> 3;
  int s = p % NS;
  int r = p / NS;

  const float* rr = ray + r * 8;
  float ox = rr[0], oy = rr[1], oz = rr[2];
  float dx = rr[3], dy = rr[4], dz = rr[5];
  float nr = rr[6], fr = rr[7];
  float depth = fulld ? fulld[r * NS + s]
                      : nr + (fr - nr) * ((s + 0.5f) / (float)NS);
  float px = ox + dx * depth, py = oy + dy * depth, pz = oz + dz * depth;
  float dinv = rsqrtf(dx * dx + dy * dy + dz * dz);
  float rdx = -dx * dinv, rdy = -dy * dinv, rdz = -dz * dinv;

  // projection
  const float* Pv = Pm + v * 12;
  float pc0 = Pv[0] * px + Pv[1] * py + Pv[2] * pz + Pv[3];
  float pc1 = Pv[4] * px + Pv[5] * py + Pv[6] * pz + Pv[7];
  float pc2 = Pv[8] * px + Pv[9] * py + Pv[10] * pz + Pv[11];
  float iz = 1.0f / (pc2 + 1e-6f);
  float whx = shape[v * 2 + 1], why = shape[v * 2 + 0];
  float gx = 2.0f * (pc0 * iz / whx) - 1.0f;
  float gy = 2.0f * (pc1 * iz / why) - 1.0f;

  // bilinear taps (border-zero)
  float xf = (gx + 1.0f) * (IW * 0.5f) - 0.5f;
  float yf = (gy + 1.0f) * (IH * 0.5f) - 0.5f;
  float x0f = floorf(xf), y0f = floorf(yf);
  float wx = xf - x0f, wy = yf - y0f;
  float xq[2] = {x0f, x0f + 1.f}, yq[2] = {y0f, y0f + 1.f};
  float wxq[2] = {1.f - wx, wx}, wyq[2] = {1.f - wy, wy};
  int xi[2], yi[2];
  bool vx[2], vy[2];
#pragma unroll
  for (int a = 0; a < 2; ++a) {
    vx[a] = (xq[a] >= 0.f) && (xq[a] <= (float)(IW - 1));
    vy[a] = (yq[a] >= 0.f) && (yq[a] <= (float)(IH - 1));
    xi[a] = (int)fminf(fmaxf(xq[a], 0.f), (float)(IW - 1));
    yi[a] = (int)fminf(fmaxf(yq[a], 0.f), (float)(IH - 1));
  }
  float wv[4];
  int off[4];
#pragma unroll
  for (int t = 0; t < 4; ++t) {
    int a = t & 1, b = t >> 1;
    wv[t] = (vx[a] && vy[b]) ? wxq[a] * wyq[b] : 0.f;
    off[t] = yi[b] * IW + xi[a];
  }
  auto samp = [&](const float* plane) -> float {
    return plane[off[0]] * wv[0] + plane[off[1]] * wv[1] +
           plane[off[2]] * wv[2] + plane[off[3]] * wv[3];
  };

  const size_t HW = (size_t)IH * IW;
  __builtin_prefetch(featT + ((size_t)(v * 2 * LF + fs) * HW), 0, 1);

  maskv[row] = samp(maskT + (size_t)v * HW);

  _Float16 xr[KPAD];
#pragma unroll
  for (int c = 0; c < 3; ++c) {
    float val = samp(imgT + ((size_t)(v * 3 + c)) * HW);
    rawrgb[(size_t)row * 3 + c] = val;
    xr[c] = (_Float16)val;
  }
#pragma unroll
  for (int c = 0; c < LF; ++c)
    xr[3 + c] = (_Float16)samp(featT + ((size_t)(v * 2 * LF + fs + c)) * HW);
  // source view direction
  float cx = campos[v * 3 + 0] - px, cy = campos[v * 3 + 1] - py,
        cz = campos[v * 3 + 2] - pz;
  float cinv = rsqrtf(cx * cx + cy * cy + cz * cz);
  xr[35] = (_Float16)(cx * cinv);
  xr[36] = (_Float16)(cy * cinv);
  xr[37] = (_Float16)(cz * cinv);
  xr[38] = (_Float16)rdx;
  xr[39] = (_Float16)rdy;
  xr[40] = (_Float16)rdz;
#pragma unroll
  for (int c = 41; c < KPAD; ++c) xr[c] = (_Float16)0.0f;

  v8h* dst = reinterpret_cast<v8h*>(X + (size_t)row * KPAD);
#pragma unroll
  for (int i = 0; i < 8; ++i) {
    v8h tmp;
#pragma unroll
    for (int j = 0; j < 8; ++j) tmp[j] = xr[i * 8 + j];
    dst[i] = tmp;
  }
}

// ------------------------------------------------------------------
// WMMA A/B fragment load per documented 16-bit layout:
// lanes 0-15 hold K {0..7,16..23}, lanes 16-31 hold K {8..15,24..31}
// (caller passes base already offset by 0 / 8 halves).
// ------------------------------------------------------------------
__device__ inline v16h ibr_frag(const _Float16* base) {
  v8h lo = *reinterpret_cast<const v8h*>(base);
  v8h hi = *reinterpret_cast<const v8h*>(base + 16);
  v16h a;
#pragma unroll
  for (int i = 0; i < 8; ++i) { a[i] = lo[i]; a[i + 8] = hi[i]; }
  return a;
}

// ------------------------------------------------------------------
// MLP + view pooling + density/blend heads.
// 128 threads = 4 waves, 1 wave = one 16-row tile = 2 points x 8 views.
// ------------------------------------------------------------------
__global__ __launch_bounds__(128) void ibr_mlp(
    const _Float16* __restrict__ X, const float* __restrict__ rawrgb,
    const float* __restrict__ maskv, const _Float16* __restrict__ W1t,
    const float* __restrict__ b1, const _Float16* __restrict__ W2t,
    const float* __restrict__ b2, const float* __restrict__ Wd,
    const float* __restrict__ bd, const float* __restrict__ Wb,
    const float* __restrict__ bb, float* __restrict__ dens,
    float* __restrict__ rgbp) {
  __shared__ __align__(16) _Float16 lds[4][16][KPAD];
  int lane = threadIdx.x & 31;
  int wave = threadIdx.x >> 5;
  int tile = blockIdx.x * 4 + wave;
  size_t m0 = (size_t)tile * 16;
  int lrow = lane & 15;            // A-matrix row within tile
  int khalf = (lane < 16) ? 0 : 8; // K-half selector
  int coln = lane & 15;            // B column / C column within N-tile

  // ---- layer 1: X(16x64) @ W1(64x64) ----
  v8f acc[4] = {};
  const _Float16* xbase = X + m0 * KPAD;
#pragma unroll
  for (int kb = 0; kb < 2; ++kb) {
    v16h a = ibr_frag(xbase + (size_t)lrow * KPAD + kb * 32 + khalf);
#pragma unroll
    for (int nt = 0; nt < 4; ++nt) {
      v16h bf = ibr_frag(W1t + (size_t)(nt * 16 + coln) * KPAD + kb * 32 + khalf);
      acc[nt] = __builtin_amdgcn_wmma_f32_16x16x32_f16(
          false, a, false, bf, (short)0, acc[nt], false, false);
    }
  }
  // bias + relu, transpose C-layout -> A-layout via LDS
#pragma unroll
  for (int nt = 0; nt < 4; ++nt) {
    float bv = b1[nt * 16 + coln];
#pragma unroll
    for (int rI = 0; rI < 8; ++rI) {
      float h = acc[nt][rI] + bv;
      h = h > 0.f ? h : 0.f;
      int mrow = (lane < 16) ? rI : (8 + rI);
      lds[wave][mrow][nt * 16 + coln] = (_Float16)h;
    }
  }
  __syncthreads();

  // ---- layer 2: H1(16x64) @ W2(64x64) ----
  v8f acc2[4] = {};
#pragma unroll
  for (int kb = 0; kb < 2; ++kb) {
    v16h a = ibr_frag(&lds[wave][lrow][kb * 32 + khalf]);
#pragma unroll
    for (int nt = 0; nt < 4; ++nt) {
      v16h bf = ibr_frag(W2t + (size_t)(nt * 16 + coln) * KPAD + kb * 32 + khalf);
      acc2[nt] = __builtin_amdgcn_wmma_f32_16x16x32_f16(
          false, a, false, bf, (short)0, acc2[nt], false, false);
    }
  }
  float h2[4][8];
#pragma unroll
  for (int nt = 0; nt < 4; ++nt) {
    float bv = b2[nt * 16 + coln];
#pragma unroll
    for (int rI = 0; rI < 8; ++rI) {
      float h = acc2[nt][rI] + bv;
      h2[nt][rI] = h > 0.f ? h : 0.f;
    }
  }

  // ---- view pooling (mean/var over the 8 accumulator rows) + density ----
  float densPart = 0.f;
#pragma unroll
  for (int nt = 0; nt < 4; ++nt) {
    float s = 0.f, s2 = 0.f;
#pragma unroll
    for (int rI = 0; rI < 8; ++rI) { s += h2[nt][rI]; s2 += h2[nt][rI] * h2[nt][rI]; }
    float mean = s * 0.125f;
    float var = s2 * 0.125f - mean * mean;
    int col = nt * 16 + coln;
    densPart += mean * Wd[col] + var * Wd[HID + col];
  }
#pragma unroll
  for (int m = 1; m < 16; m <<= 1) densPart += __shfl_xor(densPart, m, 32);
  float density = densPart + bd[0];

  // ---- blend logits + softmax over views ----
  float lg[8];
#pragma unroll
  for (int rI = 0; rI < 8; ++rI) {
    float t = 0.f;
#pragma unroll
    for (int nt = 0; nt < 4; ++nt) t += h2[nt][rI] * Wb[nt * 16 + coln];
    lg[rI] = t;
  }
#pragma unroll
  for (int rI = 0; rI < 8; ++rI) {
#pragma unroll
    for (int m = 1; m < 16; m <<= 1) lg[rI] += __shfl_xor(lg[rI], m, 32);
    lg[rI] += bb[0];
  }
  float mx = lg[0];
#pragma unroll
  for (int rI = 1; rI < 8; ++rI) mx = fmaxf(mx, lg[rI]);
  float se = 0.f;
#pragma unroll
  for (int rI = 0; rI < 8; ++rI) { lg[rI] = __expf(lg[rI] - mx); se += lg[rI]; }
  float inv = 1.0f / se;
#pragma unroll
  for (int rI = 0; rI < 8; ++rI) lg[rI] *= inv;

  // ---- hull + outputs (point = lane half) ----
  size_t pbase = (size_t)tile * 2 + ((lane < 16) ? 0 : 1);
  float msum = 0.f;
#pragma unroll
  for (int vv = 0; vv < 8; ++vv) msum += maskv[pbase * 8 + vv];
  bool hull = msum > ((float)NV - 0.001f);
  if ((lane & 15) == 0) dens[pbase] = hull ? density : 0.0f;
  int c = lane & 15;
  if (c < 3) {
    float accc = 0.f;
#pragma unroll
    for (int vv = 0; vv < 8; ++vv)
      accc += lg[vv] * rawrgb[(pbase * 8 + vv) * 3 + c];
    rgbp[pbase * 3 + c] = hull ? accc : 0.0f;
  }
}

// ------------------------------------------------------------------
// Composite (coarse) + importance sampling + sorted merge -> fulld
// ------------------------------------------------------------------
__global__ void ibr_composite_coarse(const float* __restrict__ ray,
                                     const float* __restrict__ dens,
                                     const float* __restrict__ rgbp,
                                     float* __restrict__ outp,
                                     float* __restrict__ fulld) {
  int r = blockIdx.x * blockDim.x + threadIdx.x;
  if (r >= NR) return;
  float nr = ray[r * 8 + 6], fr = ray[r * 8 + 7];
  float depth[NC], w[NC];
  for (int s = 0; s < NC; ++s)
    depth[s] = nr + (fr - nr) * ((s + 0.5f) / (float)NC);
  float T = 1.0f, crgb0 = 0, crgb1 = 0, crgb2 = 0, cdep = 0, calp = 0;
  for (int s = 0; s < NC; ++s) {
    float delta = (s < NC - 1) ? depth[s + 1] - depth[s] : 1e10f;
    float dd = dens[r * NC + s];
    dd = dd > 0.f ? dd : 0.f;
    float alpha = 1.0f - __expf(-dd * delta);
    float ws = alpha * T;
    T *= (1.0f - alpha + 1e-10f);
    w[s] = ws;
    crgb0 += ws * rgbp[((size_t)r * NC + s) * 3 + 0];
    crgb1 += ws * rgbp[((size_t)r * NC + s) * 3 + 1];
    crgb2 += ws * rgbp[((size_t)r * NC + s) * 3 + 2];
    cdep += ws * depth[s];
    calp += ws;
  }
  outp[r * 3 + 0] = crgb0;
  outp[r * 3 + 1] = crgb1;
  outp[r * 3 + 2] = crgb2;
  outp[3 * NR + r] = cdep;
  outp[4 * NR + r] = calp;

  // _sample_pdf over mids / w[1:-1]
  float cdf[NC - 1];
  cdf[0] = 0.f;
  float wsum = 0.f, pw[NC - 2];
  for (int i = 0; i < NC - 2; ++i) { pw[i] = w[i + 1] + 1e-5f; wsum += pw[i]; }
  float invs = 1.0f / wsum;
  for (int i = 0; i < NC - 2; ++i) cdf[i + 1] = cdf[i] + pw[i] * invs;
  float fd[NC];
  for (int j = 0; j < NC; ++j) {
    float u = (j + 0.5f) / (float)NC;
    int ind = NC - 1;
    for (int i = 0; i < NC - 1; ++i) {
      if (cdf[i] > u) { ind = i; break; }
    }
    int below = ind - 1;
    below = below < 0 ? 0 : (below > NC - 2 ? NC - 2 : below);
    int above = ind > NC - 2 ? NC - 2 : ind;
    float c0 = cdf[below], c1 = cdf[above];
    float bin0 = 0.5f * (depth[below + 1] + depth[below]);
    float bin1 = 0.5f * (depth[above + 1] + depth[above]);
    float dn = (c1 - c0 < 1e-5f) ? 1.0f : (c1 - c0);
    float t = (u - c0) / dn;
    fd[j] = bin0 + t * (bin1 - bin0);
  }
  // merge two sorted lists -> full depth (128)
  int i = 0, j = 0;
  float* out = fulld + (size_t)r * NSF;
  for (int k = 0; k < NSF; ++k) {
    bool takeC = (j >= NC) || (i < NC && depth[i] <= fd[j]);
    out[k] = takeC ? depth[i++] : fd[j++];
  }
}

__global__ void ibr_composite_fine(const float* __restrict__ fulld,
                                   const float* __restrict__ dens,
                                   const float* __restrict__ rgbp,
                                   float* __restrict__ outp) {
  int r = blockIdx.x * blockDim.x + threadIdx.x;
  if (r >= NR) return;
  const float* dp = fulld + (size_t)r * NSF;
  float T = 1.0f, rgb0 = 0, rgb1 = 0, rgb2 = 0, dep = 0, alp = 0;
  for (int s = 0; s < NSF; ++s) {
    float d0 = dp[s];
    float delta = (s < NSF - 1) ? dp[s + 1] - d0 : 1e10f;
    float dd = dens[(size_t)r * NSF + s];
    dd = dd > 0.f ? dd : 0.f;
    float alpha = 1.0f - __expf(-dd * delta);
    float ws = alpha * T;
    T *= (1.0f - alpha + 1e-10f);
    rgb0 += ws * rgbp[((size_t)r * NSF + s) * 3 + 0];
    rgb1 += ws * rgbp[((size_t)r * NSF + s) * 3 + 1];
    rgb2 += ws * rgbp[((size_t)r * NSF + s) * 3 + 2];
    dep += ws * d0;
    alp += ws;
  }
  outp[5 * NR + r * 3 + 0] = rgb0;
  outp[5 * NR + r * 3 + 1] = rgb1;
  outp[5 * NR + r * 3 + 2] = rgb2;
  outp[8 * NR + r] = dep;
  outp[9 * NR + r] = alp;
}

// ------------------------------------------------------------------
extern "C" void kernel_launch(void* const* d_in, const int* in_sizes, int n_in,
                              void* d_out, int out_size, void* d_ws,
                              size_t ws_size, hipStream_t stream) {
  const float* ray    = (const float*)d_in[0];
  const float* Kin    = (const float*)d_in[1];
  const float* Ein    = (const float*)d_in[2];
  const float* campos = (const float*)d_in[3];
  const float* shape  = (const float*)d_in[4];
  const float* maskT  = (const float*)d_in[5];
  const float* imgT   = (const float*)d_in[6];
  const float* featT  = (const float*)d_in[7];
  const float* cW1 = (const float*)d_in[8],  *cb1 = (const float*)d_in[9];
  const float* cW2 = (const float*)d_in[10], *cb2 = (const float*)d_in[11];
  const float* cWd = (const float*)d_in[12], *cbd = (const float*)d_in[13];
  const float* cWb = (const float*)d_in[14], *cbb = (const float*)d_in[15];
  const float* fW1 = (const float*)d_in[16], *fb1 = (const float*)d_in[17];
  const float* fW2 = (const float*)d_in[18], *fb2 = (const float*)d_in[19];
  const float* fWd = (const float*)d_in[20], *fbd = (const float*)d_in[21];
  const float* fWb = (const float*)d_in[22], *fbb = (const float*)d_in[23];
  (void)in_sizes; (void)n_in; (void)out_size; (void)ws_size;

  // workspace carve (sized for the larger fine stage, reused for coarse)
  const size_t maxRows = (size_t)NR * NSF * NV; // 1,048,576
  const size_t maxPts  = (size_t)NR * NSF;      // 131,072
  size_t off = 0;
  auto carve = [&](size_t bytes) -> char* {
    char* p = (char*)d_ws + off;
    off += (bytes + 255) & ~(size_t)255;
    return p;
  };
  _Float16* X      = (_Float16*)carve(maxRows * KPAD * sizeof(_Float16));
  float* rawrgb    = (float*)carve(maxRows * 3 * sizeof(float));
  float* maskv     = (float*)carve(maxRows * sizeof(float));
  float* densBuf   = (float*)carve(maxPts * sizeof(float));
  float* rgbpBuf   = (float*)carve(maxPts * 3 * sizeof(float));
  float* fulld     = (float*)carve((size_t)NR * NSF * sizeof(float));
  float* Pm        = (float*)carve(NV * 12 * sizeof(float));
  _Float16* cW1t   = (_Float16*)carve(HID * KPAD * sizeof(_Float16));
  _Float16* cW2t   = (_Float16*)carve(HID * KPAD * sizeof(_Float16));
  _Float16* fW1t   = (_Float16*)carve(HID * KPAD * sizeof(_Float16));
  _Float16* fW2t   = (_Float16*)carve(HID * KPAD * sizeof(_Float16));
  float* outp = (float*)d_out;

  ibr_prep_pmat<<<1, 128, 0, stream>>>(Kin, Ein, Pm);
  ibr_prep_w<<<16, 256, 0, stream>>>(cW1, cW1t, 41);
  ibr_prep_w<<<16, 256, 0, stream>>>(cW2, cW2t, 64);
  ibr_prep_w<<<16, 256, 0, stream>>>(fW1, fW1t, 41);
  ibr_prep_w<<<16, 256, 0, stream>>>(fW2, fW2t, 64);

  // ---- coarse ----
  int rowsC = NR * NC * NV; // 524288
  ibr_sample<<<rowsC / 256, 256, 0, stream>>>(ray, Pm, shape, campos, maskT,
                                              imgT, featT, nullptr, NC, 0, X,
                                              rawrgb, maskv);
  ibr_mlp<<<rowsC / 64, 128, 0, stream>>>(X, rawrgb, maskv, cW1t, cb1, cW2t,
                                          cb2, cWd, cbd, cWb, cbb, densBuf,
                                          rgbpBuf);
  ibr_composite_coarse<<<NR / 256, 256, 0, stream>>>(ray, densBuf, rgbpBuf,
                                                     outp, fulld);

  // ---- fine ----
  int rowsF = NR * NSF * NV; // 1048576
  ibr_sample<<<rowsF / 256, 256, 0, stream>>>(ray, Pm, shape, campos, maskT,
                                              imgT, featT, fulld, NSF, LF, X,
                                              rawrgb, maskv);
  ibr_mlp<<<rowsF / 64, 128, 0, stream>>>(X, rawrgb, maskv, fW1t, fb1, fW2t,
                                          fb2, fWd, fbd, fWb, fbb, densBuf,
                                          rgbpBuf);
  ibr_composite_fine<<<NR / 256, 256, 0, stream>>>(fulld, densBuf, rgbpBuf,
                                                   outp);
}
